// SAModule_43894565765751
// MI455X (gfx1250) — compile-verified, matrix operand order checked by reading, not run on hardware
//
#include <hip/hip_runtime.h>
#include <hip/hip_bf16.h>
#include <stdint.h>

// ---------------- problem constants ----------------
constexpr int   kB   = 8;
constexpr int   kN   = 4096;
constexpr int   kS   = 2048;
constexpr int   kK   = 64;
constexpr int   kCIN = 64;
constexpr int   kC0  = 67;     // CIN + 3
constexpr int   kC0P = 96;     // padded to 3 x 32 for bf16 WMMA k=32
constexpr int   kHID = 128;
constexpr int   kOUT = 128;
constexpr float kR2  = 0.2f * 0.2f;
constexpr float kEPS = 1e-5f;

// ---------------- WMMA types (CDNA5 wave32) ----------------
typedef __bf16 v16bf __attribute__((ext_vector_type(16)));
typedef float  v8f   __attribute__((ext_vector_type(8)));

union FragBF { v16bf v; uint4 q[2]; };

__device__ __forceinline__ unsigned short f2bf(float f) {
    union { float f; unsigned u; } v; v.f = f;
    unsigned u = v.u;
    u += 0x7fffu + ((u >> 16) & 1u);           // RNE to bf16
    return (unsigned short)(u >> 16);
}
__device__ __forceinline__ float bf2f(unsigned short h) {
    union { unsigned u; float f; } v; v.u = ((unsigned)h) << 16;
    return v.f;
}

// ---------------- CDNA5 async global->LDS copy ----------------
#if defined(__has_builtin)
#if __has_builtin(__builtin_amdgcn_global_load_async_to_lds_b128)
#define HAVE_ASYNC_LDS 1
#endif
#endif

typedef int v4i_vs __attribute__((vector_size(16)));   // matches builtin's pointee type

__device__ __forceinline__ void async_copy_b128(void* lds_dst, const void* gsrc) {
#ifdef HAVE_ASYNC_LDS
    __builtin_amdgcn_global_load_async_to_lds_b128(
        (__attribute__((address_space(1))) v4i_vs*)gsrc,
        (__attribute__((address_space(3))) v4i_vs*)lds_dst,
        0, 0);
#else
    *(uint4*)lds_dst = *(const uint4*)gsrc;
#endif
}

__device__ __forceinline__ void wait_async() {
#if defined(__has_builtin)
#if __has_builtin(__builtin_amdgcn_s_wait_asynccnt)
    __builtin_amdgcn_s_wait_asynccnt(0);
#else
    asm volatile("s_wait_asynccnt 0" ::: "memory");
#endif
#else
    asm volatile("s_wait_asynccnt 0" ::: "memory");
#endif
}

// ============================================================
// 0. zero the stats scratch (d_ws is poisoned, not re-zeroed)
// ============================================================
__global__ void init_kernel(float* stats, unsigned* cnt) {
    int t = blockIdx.x * blockDim.x + threadIdx.x;
    if (t < 4 * kHID) stats[t] = 0.f;
    if (t == 0) *cnt = 0u;
}

// ============================================================
// 0b. one-time weight prep: transpose + fp32->bf16 + k-pad.
// ============================================================
__global__ void prep_weights_kernel(const float* __restrict__ W0,
                                    const float* __restrict__ W1,
                                    const float* __restrict__ W2,
                                    unsigned short* __restrict__ w0t,
                                    unsigned short* __restrict__ w1t,
                                    unsigned short* __restrict__ w2t) {
    int t = blockIdx.x * blockDim.x + threadIdx.x;
    if (t < kHID * kC0P) {                 // w0t[n][k], zero-pad k >= 67
        int n = t / kC0P, k = t % kC0P;
        w0t[t] = f2bf(k < kC0 ? W0[k * kHID + n] : 0.f);
    }
    if (t < kHID * kHID) {                 // w1t/w2t[n][k]
        int n = t / kHID, k = t % kHID;
        w1t[t] = f2bf(W1[k * kHID + n]);
        w2t[t] = f2bf(W2[k * kHID + n]);
    }
}

// ============================================================
// 1. FPS: one block per batch, 4096 points in registers
// ============================================================
__global__ void __launch_bounds__(1024) fps_kernel(const float* __restrict__ pos,
                                                   int* __restrict__ idx,
                                                   float* __restrict__ ctr) {
    constexpr int NTH = 1024, PT = kN / NTH;
    const int b = blockIdx.x;
    const int t = threadIdx.x;
    const float* p = pos + (size_t)b * kN * 3;

    float px[PT], py[PT], pz[PT], mind[PT];
#pragma unroll
    for (int i = 0; i < PT; ++i) {
        int pt = t + i * NTH;
        px[i] = p[pt * 3 + 0];
        py[i] = p[pt * 3 + 1];
        pz[i] = p[pt * 3 + 2];
        mind[i] = 3.4e38f;
    }

    __shared__ float selx, sely, selz;
    __shared__ int   sLast;
    __shared__ float wv[32];
    __shared__ int   wi[32];

    if (t == 0) { selx = p[0]; sely = p[1]; selz = p[2]; }
    __syncthreads();

    int last = 0;
    const int wave = t >> 5, lane = t & 31;
    for (int s = 0; s < kS; ++s) {
        if (t == 0) {
            idx[b * kS + s] = last;
            ctr[((size_t)b * kS + s) * 3 + 0] = selx;
            ctr[((size_t)b * kS + s) * 3 + 1] = sely;
            ctr[((size_t)b * kS + s) * 3 + 2] = selz;
        }
        const float cx = selx, cy = sely, cz = selz;
        float bestv = -1.f; int besti = 0;
#pragma unroll
        for (int i = 0; i < PT; ++i) {
            float dx = px[i] - cx, dy = py[i] - cy, dz = pz[i] - cz;
            float d = dx * dx + dy * dy + dz * dz;
            mind[i] = fminf(mind[i], d);
            if (mind[i] > bestv) { bestv = mind[i]; besti = t + i * NTH; }
        }
        // wave32 argmax reduce
#pragma unroll
        for (int off = 16; off > 0; off >>= 1) {
            float ov = __shfl_xor(bestv, off, 32);
            int   oi = __shfl_xor(besti, off, 32);
            if (ov > bestv) { bestv = ov; besti = oi; }
        }
        if (lane == 0) { wv[wave] = bestv; wi[wave] = besti; }
        __syncthreads();
        if (wave == 0) {
            bestv = wv[lane]; besti = wi[lane];
#pragma unroll
            for (int off = 16; off > 0; off >>= 1) {
                float ov = __shfl_xor(bestv, off, 32);
                int   oi = __shfl_xor(besti, off, 32);
                if (ov > bestv) { bestv = ov; besti = oi; }
            }
            if (lane == 0) sLast = besti;
        }
        __syncthreads();
        last = sLast;
        int slot = last >> 10, owner = last & 1023;
        if (t == owner) { selx = px[slot]; sely = py[slot]; selz = pz[slot]; }
        __syncthreads();
    }
}

// ============================================================
// 2. radius neighbors: one wave per centroid, deterministic
//    ballot/popcount compaction of first K in-radius points
// ============================================================
__global__ void __launch_bounds__(256) nbr_kernel(const float* __restrict__ pos,
                                                  const float* __restrict__ ctr,
                                                  int* __restrict__ nbr,
                                                  unsigned* __restrict__ edgecnt) {
    const int wave = threadIdx.x >> 5, lane = threadIdx.x & 31;
    const int c = blockIdx.x * 8 + wave;
    const int b = c / kS;
    const float* p = pos + (size_t)b * kN * 3;
    const float cx = ctr[(size_t)c * 3 + 0];
    const float cy = ctr[(size_t)c * 3 + 1];
    const float cz = ctr[(size_t)c * 3 + 2];
    int* nb = nbr + (size_t)c * kK;

    for (int j = lane; j < kK; j += 32) nb[j] = -1;

    int cnt = 0;
    for (int base = 0; base < kN; base += 32) {
        int pi = base + lane;
        float dx = p[pi * 3 + 0] - cx;
        float dy = p[pi * 3 + 1] - cy;
        float dz = p[pi * 3 + 2] - cz;
        bool in = (dx * dx + dy * dy + dz * dz) <= kR2;
        unsigned mm = (unsigned)__ballot(in);
        int below = __popc(mm & ((1u << lane) - 1u));
        if (in) {
            int slot = cnt + below;
            if (slot < kK) nb[slot] = pi;
        }
        cnt += __popc(mm);
        if (cnt >= kK) break;
    }
    if (lane == 0) atomicAdd(edgecnt, (unsigned)min(cnt, kK));
}

// ============================================================
// 3. GEMM0: fused gather + [64 x 96] @ [96 x 128] bf16 WMMA
//    + global BN statistics (masked). Weights via async DMA.
// ============================================================
__global__ void __launch_bounds__(128) gemm0_kernel(const float* __restrict__ x,
                                                    const float* __restrict__ pos,
                                                    const unsigned short* __restrict__ w0t,
                                                    const float* __restrict__ b0,
                                                    const float* __restrict__ ctr,
                                                    const int* __restrict__ nbr,
                                                    unsigned short* __restrict__ h0,
                                                    float* __restrict__ gsum,
                                                    float* __restrict__ gsq) {
    __shared__ __align__(16) unsigned short sFeat[kK * kC0P];  // [row][k]  12 KB
    __shared__ __align__(16) unsigned short sWt[kHID * kC0P];  // [n][k]    24 KB
    __shared__ int   sNbr[kK];
    __shared__ float sSum[kHID], sSq[kHID];

    const int c = blockIdx.x;
    const int b = c / kS;
    const int t = threadIdx.x;
    const int lane = t & 31, wave = t >> 5;

    if (t == 0) __builtin_prefetch(w0t, 0, 0);     // global_prefetch_b8
    if (t < kK)  sNbr[t] = nbr[(size_t)c * kK + t];
    if (t < kHID) { sSum[t] = 0.f; sSq[t] = 0.f; }

    // async-DMA pre-converted bf16 W0^T into LDS (overlaps gather below)
    for (int j = t; j < (kHID * kC0P) / 8; j += 128)
        async_copy_b128(&sWt[j * 8], &w0t[j * 8]);
    __syncthreads();   // sNbr visible

    const float cx = ctr[(size_t)c * 3 + 0];
    const float cy = ctr[(size_t)c * 3 + 1];
    const float cz = ctr[(size_t)c * 3 + 2];
    for (int j = t; j < kK * kC0P; j += 128) {     // gather + concat + pad
        int r = j / kC0P, k = j % kC0P;
        int pi = sNbr[r];
        float v = 0.f;
        if (pi >= 0) {
            if (k < kCIN)      v = x[((size_t)b * kN + pi) * kCIN + k];
            else if (k == 64)  v = pos[((size_t)b * kN + pi) * 3 + 0] - cx;
            else if (k == 65)  v = pos[((size_t)b * kN + pi) * 3 + 1] - cy;
            else if (k == 66)  v = pos[((size_t)b * kN + pi) * 3 + 2] - cz;
        }
        sFeat[j] = f2bf(v);
    }
    wait_async();
    __syncthreads();

    v8f acc[8];
    v8f zero = {0.f, 0.f, 0.f, 0.f, 0.f, 0.f, 0.f, 0.f};
#pragma unroll
    for (int tn = 0; tn < 8; ++tn) acc[tn] = zero;

    const int mrow = wave * 16 + (lane & 15);
#pragma unroll
    for (int kc = 0; kc < 3; ++kc) {
        const int kbase = kc * 32 + ((lane >= 16) ? 8 : 0);
        FragBF fa;
        const uint4* pa = reinterpret_cast<const uint4*>(&sFeat[mrow * kC0P + kbase]);
        fa.q[0] = pa[0]; fa.q[1] = pa[2];
#pragma unroll
        for (int tn = 0; tn < 8; ++tn) {
            const int coln = tn * 16 + (lane & 15);
            FragBF fb;
            const uint4* pb = reinterpret_cast<const uint4*>(&sWt[coln * kC0P + kbase]);
            fb.q[0] = pb[0]; fb.q[1] = pb[2];
            acc[tn] = __builtin_amdgcn_wmma_f32_16x16x32_bf16(
                false, fa.v, false, fb.v, (short)0, acc[tn], false, false);
        }
    }

#pragma unroll
    for (int tn = 0; tn < 8; ++tn) {
        const int coln = tn * 16 + (lane & 15);
        const float bias = b0[coln];
#pragma unroll
        for (int r = 0; r < 8; ++r) {
            int row = wave * 16 + ((lane < 16) ? r : 8 + r);
            float val = acc[tn][r] + bias;
            h0[((size_t)c * kK + row) * kHID + coln] = f2bf(val);
            if (sNbr[row] >= 0) {
                atomicAdd(&sSum[coln], val);
                atomicAdd(&sSq[coln],  val * val);
            }
        }
    }
    __syncthreads();
    if (t < kHID) {
        atomicAdd(&gsum[t], sSum[t]);
        atomicAdd(&gsq[t],  sSq[t]);
    }
}

// ============================================================
// 3b/4b. BN finalize: mean/var -> affine (scale, shift)
// ============================================================
__global__ void bn_finalize_kernel(const float* __restrict__ gsum,
                                   const float* __restrict__ gsq,
                                   const unsigned* __restrict__ cnt,
                                   const float* __restrict__ g,
                                   const float* __restrict__ be,
                                   float* __restrict__ bnp) {
    int t = threadIdx.x;
    if (t < kHID) {
        float n = (float)(*cnt);
        float mean = gsum[t] / n;
        float var  = gsq[t] / n - mean * mean;
        float a = g[t] * rsqrtf(var + kEPS);
        bnp[t]        = a;
        bnp[kHID + t] = be[t] - mean * a;
    }
}

// ============================================================
// 4. GEMM1: bn+relu(h0) @ W1, bf16 WMMA, + BN stats
// ============================================================
__global__ void __launch_bounds__(128) gemm1_kernel(const unsigned short* __restrict__ h0,
                                                    const unsigned short* __restrict__ w1t,
                                                    const float* __restrict__ b1,
                                                    const float* __restrict__ bnp,
                                                    const int* __restrict__ nbr,
                                                    unsigned short* __restrict__ h1,
                                                    float* __restrict__ gsum,
                                                    float* __restrict__ gsq) {
    __shared__ __align__(16) unsigned short sAct[kK * kHID];   // 16 KB
    __shared__ __align__(16) unsigned short sWt[kHID * kHID];  // 32 KB
    __shared__ int   sNbr[kK];
    __shared__ float sSum[kHID], sSq[kHID];

    const int c = blockIdx.x;
    const int t = threadIdx.x;
    const int lane = t & 31, wave = t >> 5;

    if (t < kK)  sNbr[t] = nbr[(size_t)c * kK + t];
    if (t < kHID) { sSum[t] = 0.f; sSq[t] = 0.f; }

    // async-DMA bf16 W1^T into LDS (overlaps BN+ReLU staging below)
    for (int j = t; j < (kHID * kHID) / 8; j += 128)
        async_copy_b128(&sWt[j * 8], &w1t[j * 8]);

    for (int j = t; j < kK * kHID; j += 128) {
        int r = j / kHID, k = j % kHID;
        float hv = bf2f(h0[((size_t)c * kK + r) * kHID + k]);
        float a = fmaf(hv, bnp[k], bnp[kHID + k]);
        sAct[j] = f2bf(a > 0.f ? a : 0.f);
    }
    wait_async();
    __syncthreads();

    v8f acc[8];
    v8f zero = {0.f, 0.f, 0.f, 0.f, 0.f, 0.f, 0.f, 0.f};
#pragma unroll
    for (int tn = 0; tn < 8; ++tn) acc[tn] = zero;

    const int mrow = wave * 16 + (lane & 15);
#pragma unroll
    for (int kc = 0; kc < 4; ++kc) {
        const int kbase = kc * 32 + ((lane >= 16) ? 8 : 0);
        FragBF fa;
        const uint4* pa = reinterpret_cast<const uint4*>(&sAct[mrow * kHID + kbase]);
        fa.q[0] = pa[0]; fa.q[1] = pa[2];
#pragma unroll
        for (int tn = 0; tn < 8; ++tn) {
            const int coln = tn * 16 + (lane & 15);
            FragBF fb;
            const uint4* pb = reinterpret_cast<const uint4*>(&sWt[coln * kHID + kbase]);
            fb.q[0] = pb[0]; fb.q[1] = pb[2];
            acc[tn] = __builtin_amdgcn_wmma_f32_16x16x32_bf16(
                false, fa.v, false, fb.v, (short)0, acc[tn], false, false);
        }
    }

#pragma unroll
    for (int tn = 0; tn < 8; ++tn) {
        const int coln = tn * 16 + (lane & 15);
        const float bias = b1[coln];
#pragma unroll
        for (int r = 0; r < 8; ++r) {
            int row = wave * 16 + ((lane < 16) ? r : 8 + r);
            float val = acc[tn][r] + bias;
            h1[((size_t)c * kK + row) * kHID + coln] = f2bf(val);
            if (sNbr[row] >= 0) {
                atomicAdd(&sSum[coln], val);
                atomicAdd(&sSq[coln],  val * val);
            }
        }
    }
    __syncthreads();
    if (t < kHID) {
        atomicAdd(&gsum[t], sSum[t]);
        atomicAdd(&gsq[t],  sSq[t]);
    }
}

// ============================================================
// 5. GEMM2: bn+relu(h1) @ W2 + b2, masked max over K -> out
// ============================================================
__global__ void __launch_bounds__(128) gemm2_kernel(const unsigned short* __restrict__ h1,
                                                    const unsigned short* __restrict__ w2t,
                                                    const float* __restrict__ b2,
                                                    const float* __restrict__ bnp,
                                                    const int* __restrict__ nbr,
                                                    const float* __restrict__ ctr,
                                                    float* __restrict__ out,
                                                    float* __restrict__ outctr,
                                                    float* __restrict__ outbatch) {
    __shared__ __align__(16) unsigned short sAct[kK * kHID];
    __shared__ __align__(16) unsigned short sWt[kHID * kHID];
    __shared__ int   sNbr[kK];
    __shared__ float sMax[4 * kHID];

    const int c = blockIdx.x;
    const int t = threadIdx.x;
    const int lane = t & 31, wave = t >> 5;

    if (t < kK) sNbr[t] = nbr[(size_t)c * kK + t];

    for (int j = t; j < (kHID * kHID) / 8; j += 128)
        async_copy_b128(&sWt[j * 8], &w2t[j * 8]);

    for (int j = t; j < kK * kHID; j += 128) {
        int r = j / kHID, k = j % kHID;
        float hv = bf2f(h1[((size_t)c * kK + r) * kHID + k]);
        float a = fmaf(hv, bnp[k], bnp[kHID + k]);
        sAct[j] = f2bf(a > 0.f ? a : 0.f);
    }
    wait_async();
    __syncthreads();

    v8f acc[8];
    v8f zero = {0.f, 0.f, 0.f, 0.f, 0.f, 0.f, 0.f, 0.f};
#pragma unroll
    for (int tn = 0; tn < 8; ++tn) acc[tn] = zero;

    const int mrow = wave * 16 + (lane & 15);
#pragma unroll
    for (int kc = 0; kc < 4; ++kc) {
        const int kbase = kc * 32 + ((lane >= 16) ? 8 : 0);
        FragBF fa;
        const uint4* pa = reinterpret_cast<const uint4*>(&sAct[mrow * kHID + kbase]);
        fa.q[0] = pa[0]; fa.q[1] = pa[2];
#pragma unroll
        for (int tn = 0; tn < 8; ++tn) {
            const int coln = tn * 16 + (lane & 15);
            FragBF fb;
            const uint4* pb = reinterpret_cast<const uint4*>(&sWt[coln * kHID + kbase]);
            fb.q[0] = pb[0]; fb.q[1] = pb[2];
            acc[tn] = __builtin_amdgcn_wmma_f32_16x16x32_bf16(
                false, fa.v, false, fb.v, (short)0, acc[tn], false, false);
        }
    }

    // masked max over this wave's 16 rows, per column
#pragma unroll
    for (int tn = 0; tn < 8; ++tn) {
        const int coln = tn * 16 + (lane & 15);
        const float bias = b2[coln];
        float m = -3.402823e38f;
#pragma unroll
        for (int r = 0; r < 8; ++r) {
            int row = wave * 16 + ((lane < 16) ? r : 8 + r);
            if (sNbr[row] >= 0) m = fmaxf(m, acc[tn][r] + bias);
        }
        float o = __shfl_xor(m, 16, 32);       // fold rows 8..15 into 0..7 half
        m = fmaxf(m, o);
        if (lane < 16) sMax[wave * kHID + tn * 16 + lane] = m;
    }
    __syncthreads();
    if (t < kHID) {
        float m = fmaxf(fmaxf(sMax[t], sMax[kHID + t]),
                        fmaxf(sMax[2 * kHID + t], sMax[3 * kHID + t]));
        out[(size_t)c * kOUT + t] = m;
    }
    if (t < 3)  outctr[(size_t)c * 3 + t] = ctr[(size_t)c * 3 + t];
    if (t == 3) outbatch[c] = (float)(c / kS);
}

// ============================================================
// host-side launcher
// ============================================================
extern "C" void kernel_launch(void* const* d_in, const int* in_sizes, int n_in,
                              void* d_out, int out_size, void* d_ws, size_t ws_size,
                              hipStream_t stream) {
    (void)in_sizes; (void)n_in; (void)out_size; (void)ws_size;

    const float* x   = (const float*)d_in[0];
    const float* pos = (const float*)d_in[1];
    const float* W0  = (const float*)d_in[3];
    const float* b0  = (const float*)d_in[4];
    const float* g0  = (const float*)d_in[5];
    const float* be0 = (const float*)d_in[6];
    const float* W1  = (const float*)d_in[7];
    const float* b1  = (const float*)d_in[8];
    const float* g1  = (const float*)d_in[9];
    const float* be1 = (const float*)d_in[10];
    const float* W2  = (const float*)d_in[11];
    const float* b2  = (const float*)d_in[12];

    char* ws = (char*)d_ws;
    size_t off = 0;
    auto alloc = [&](size_t bytes) -> void* {
        void* p = ws + off;
        off += (bytes + 255) & ~(size_t)255;
        return p;
    };
    int*            idx   = (int*)alloc((size_t)kB * kS * sizeof(int));
    float*          ctr   = (float*)alloc((size_t)kB * kS * 3 * sizeof(float));
    int*            nbr   = (int*)alloc((size_t)kB * kS * kK * sizeof(int));
    float*          stats = (float*)alloc(4 * kHID * sizeof(float));
    unsigned*       cnt   = (unsigned*)alloc(sizeof(unsigned));
    float*          bnp0  = (float*)alloc(2 * kHID * sizeof(float));
    float*          bnp1  = (float*)alloc(2 * kHID * sizeof(float));
    unsigned short* w0t   = (unsigned short*)alloc((size_t)kHID * kC0P * 2);
    unsigned short* w1t   = (unsigned short*)alloc((size_t)kHID * kHID * 2);
    unsigned short* w2t   = (unsigned short*)alloc((size_t)kHID * kHID * 2);
    unsigned short* h0    = (unsigned short*)alloc((size_t)kB * kS * kK * kHID * 2);
    unsigned short* h1    = (unsigned short*)alloc((size_t)kB * kS * kK * kHID * 2);

    float* gsum0 = stats;
    float* gsq0  = stats + kHID;
    float* gsum1 = stats + 2 * kHID;
    float* gsq1  = stats + 3 * kHID;

    float* out      = (float*)d_out;
    float* outctr   = out + (size_t)kB * kS * kOUT;
    float* outbatch = outctr + (size_t)kB * kS * 3;

    init_kernel<<<2, 256, 0, stream>>>(stats, cnt);
    prep_weights_kernel<<<(kHID * kHID + 255) / 256, 256, 0, stream>>>(W0, W1, W2, w0t, w1t, w2t);
    fps_kernel<<<kB, 1024, 0, stream>>>(pos, idx, ctr);
    nbr_kernel<<<(kB * kS) / 8, 256, 0, stream>>>(pos, ctr, nbr, cnt);
    gemm0_kernel<<<kB * kS, 128, 0, stream>>>(x, pos, w0t, b0, ctr, nbr, h0, gsum0, gsq0);
    bn_finalize_kernel<<<1, 128, 0, stream>>>(gsum0, gsq0, cnt, g0, be0, bnp0);
    gemm1_kernel<<<kB * kS, 128, 0, stream>>>(h0, w1t, b1, bnp0, nbr, h1, gsum1, gsq1);
    bn_finalize_kernel<<<1, 128, 0, stream>>>(gsum1, gsq1, cnt, g1, be1, bnp1);
    gemm2_kernel<<<kB * kS, 128, 0, stream>>>(h1, w2t, b2, bnp1, nbr, ctr, out, outctr, outbatch);
}
